// BaseRNN_73074573574760
// MI455X (gfx1250) — matrix operand order repaired and checked by reading, not compiled, not run on hardware
//
#include <hip/hip_runtime.h>
#include <hip/hip_bf16.h>
#include <math.h>

// Problem constants (from reference): B=32, S=2048, H=256, L=2
#define RB 32
#define RS 2048
#define RH 256
#define RL 2
#define HP 264   // padded LDS h-row pitch (bf16 elems): 528B = 132 dwords -> bank-conflict free b128 reads

typedef __attribute__((ext_vector_type(16))) __bf16 v16bf;
typedef __attribute__((ext_vector_type(8)))  __bf16 v8bf;
typedef __attribute__((ext_vector_type(8)))  float  v8f;

// ---------------------------------------------------------------------------
// WMMA B-fragment layout (16x16x32 bf16, wave32), per CDNA5 ISA 7.12.2:
//   B is KxN = 32x16 per k-chunk. Lane L holds column N = L%16, rows
//   K = 8*(L/16) + s for s<8, K = 16 + 8*(L/16) + (s-8) for s>=8.
// We store fragments split into lo/hi 8-slot halves so each half is a
// contiguous 16B-per-lane run (one ds_load_b128 / global_load_b128, no
// bank conflicts):
//   elem index = ((((mat*16 + nt)*8 + kc)*2 + hsel)*32 + lane)*8 + s8
//   with slot s = hsel*8 + s8,  K = kc*32 + 16*hsel + 8*(lane/16) + s8,  N = nt*16 + L%16
// mats: 0 = W_ih[0], 1 = W_hh[0], 2 = W_ih[1], 3 = W_hh[1]
// ---------------------------------------------------------------------------
__global__ void prep_weights_kernel(const float* __restrict__ W_ih,
                                    const float* __restrict__ W_hh,
                                    __bf16* __restrict__ wfrag) {
  int i = blockIdx.x * blockDim.x + threadIdx.x;   // 0 .. 4*16*8*2*32*8 - 1 = 262143
  int s8   =  i        & 7;
  int lane = (i >> 3)  & 31;
  int hsel = (i >> 8)  & 1;
  int kc   = (i >> 9)  & 7;
  int nt   = (i >> 12) & 15;
  int mat  =  i >> 16;                             // 0..3
  int l = mat >> 1;
  const float* src = ((mat & 1) ? W_hh : W_ih) + l * RH * RH;
  int n = nt * 16 + (lane & 15);
  int k = kc * 32 + hsel * 16 + (lane >> 4) * 8 + s8;
  // B[k][n] = W[n][k]  (out = in @ W^T)
  wfrag[i] = (__bf16)src[n * RH + k];
}

// ---------------------------------------------------------------------------
// Input projection: xp[m][n] = sum_k x[m][k]*W_ih[n][k] + b_ih[n] + b_hh[n]
// (b_hh folded in so the recurrence init is a pure load). One wave computes a
// 16x16 M-tile across 4 N-tiles (A fragments reused 4x). Grid covers
// M-tiles = 65536/16 = 4096, n-groups = 4.
// ---------------------------------------------------------------------------
__global__ void __launch_bounds__(256) proj_kernel(
    const float* __restrict__ x,        // [B*S, H] fp32
    const __bf16* __restrict__ wfragB,  // this layer's W_ih fragments (16*8*2*32*8)
    const float* __restrict__ b_ih,     // [H]
    const float* __restrict__ b_hh,     // [H]
    float* __restrict__ xp) {           // [B*S, H] fp32
  const int lane = threadIdx.x & 31;
  const int wave = threadIdx.x >> 5;
  const int gw   = blockIdx.x * 8 + wave;
  const int mt   = gw >> 2;             // 0..4095
  const int ng   = gw & 3;              // n-tile group (4 tiles each)
  const int half = lane >> 4;
  const int lm   = lane & 15;
  const int row  = mt * 16 + lm;        // A-matrix row for this lane

  v8f acc[4];
#pragma unroll
  for (int j = 0; j < 4; ++j) {
    int n = (ng * 4 + j) * 16 + lm;
    float bias = b_ih[n] + b_hh[n];
#pragma unroll
    for (int r = 0; r < 8; ++r) acc[j][r] = bias;
  }

#pragma unroll
  for (int kc = 0; kc < 8; ++kc) {
    // A fragment (16x32 bf16): lane L -> M=L%16, K = kc*32 + (L/16)*8 + {0..7, 16..23}
    const float* xr = x + row * RH + kc * 32 + half * 8;
    float4 f0 = *(const float4*)(xr);
    float4 f1 = *(const float4*)(xr + 4);
    float4 f2 = *(const float4*)(xr + 16);
    float4 f3 = *(const float4*)(xr + 20);
    v16bf a;
    a[0]=(__bf16)f0.x; a[1]=(__bf16)f0.y; a[2]=(__bf16)f0.z; a[3]=(__bf16)f0.w;
    a[4]=(__bf16)f1.x; a[5]=(__bf16)f1.y; a[6]=(__bf16)f1.z; a[7]=(__bf16)f1.w;
    a[8]=(__bf16)f2.x; a[9]=(__bf16)f2.y; a[10]=(__bf16)f2.z; a[11]=(__bf16)f2.w;
    a[12]=(__bf16)f3.x; a[13]=(__bf16)f3.y; a[14]=(__bf16)f3.z; a[15]=(__bf16)f3.w;
#pragma unroll
    for (int j = 0; j < 4; ++j) {
      int nt = ng * 4 + j;
      const __bf16* wp = wfragB + ((size_t)((nt * 8 + kc) * 2) * 32 + lane) * 8;
      v8bf blo = *(const v8bf*)(wp);
      v8bf bhi = *(const v8bf*)(wp + 256);
      v16bf bm = __builtin_shufflevector(blo, bhi, 0,1,2,3,4,5,6,7,8,9,10,11,12,13,14,15);
      acc[j] = __builtin_amdgcn_wmma_f32_16x16x32_bf16(
          false, a, false, bm, (short)0, acc[j], false, false);
    }
  }

#pragma unroll
  for (int j = 0; j < 4; ++j) {
    int n = (ng * 4 + j) * 16 + lm;
#pragma unroll
    for (int r = 0; r < 8; ++r) {
      int rr = mt * 16 + r + 8 * half;  // C/D layout: VGPR r -> M = r + 8*(L/16)
      xp[rr * RH + n] = acc[j][r];
    }
  }
}

// ---------------------------------------------------------------------------
// Recurrence: one persistent workgroup (32 waves = 32 output 16x16 tiles).
// W_hh fragments live in LDS (128KB), h double-buffered in LDS (2x ~17KB).
// Per step, per wave: 8x wmma_f32_16x16x32_bf16 over K=256, tanh, scatter.
// ---------------------------------------------------------------------------
__global__ void __launch_bounds__(1024) scan_kernel(
    const float* __restrict__ xp,        // [B, S, H] fp32 (x@W_ih^T + b_ih + b_hh)
    const __bf16* __restrict__ whh_frag, // 16*8*2*32*8 bf16
    const float* __restrict__ h0l,       // [B, H] fp32
    float* __restrict__ out_x,           // [B, S, H] fp32
    float* __restrict__ finals) {        // [B, H] fp32
  extern __shared__ __bf16 lds[];
  __bf16* wlds = lds;                          // 65536 bf16 = 128KB
  __bf16* hbuf = lds + 16 * 8 * 2 * 32 * 8;    // 2 * RB * HP bf16

  const int tid = threadIdx.x;

  // Stage W_hh fragments into LDS (16B chunks, coalesced)
  {
    const uint4* src = (const uint4*)whh_frag;
    uint4* dst = (uint4*)wlds;
    for (int i = tid; i < 65536 / 8; i += 1024) dst[i] = src[i];
  }
  // h <- h0 (bf16, padded pitch)
  for (int i = tid; i < RB * RH; i += 1024) {
    int b = i >> 8, n = i & 255;
    hbuf[b * HP + n] = (__bf16)h0l[i];
  }
  __syncthreads();

  const int lane = tid & 31;
  const int wave = tid >> 5;
  const int mt   = wave >> 4;   // 0..1  (batch tile)
  const int nt   = wave & 15;   // 0..15 (hidden tile)
  const int half = lane >> 4;
  const int lm   = lane & 15;
  const int n    = nt * 16 + lm;

  int cur = 0;
  for (int t = 0; t < RS; ++t) {
    __bf16* hc = hbuf + cur * (RB * HP);
    __bf16* hn = hbuf + (cur ^ 1) * (RB * HP);

    // acc <- xp[:, t, :] tile (already includes both biases)
    v8f acc;
#pragma unroll
    for (int r = 0; r < 8; ++r) {
      int b = mt * 16 + r + 8 * half;
      acc[r] = xp[(b * RS + t) * RH + n];
    }
    if (t + 1 < RS) {  // uniform branch; warm L2->WGP path for next step
      int b0 = mt * 16 + 8 * half;
      __builtin_prefetch(&xp[(b0 * RS + t + 1) * RH + n], 0, 0);
    }

#pragma unroll
    for (int kc = 0; kc < 8; ++kc) {
      const __bf16* ha = hc + (mt * 16 + lm) * HP + kc * 32 + half * 8;
      v8bf alo = *(const v8bf*)(ha);
      v8bf ahi = *(const v8bf*)(ha + 16);
      v16bf a = __builtin_shufflevector(alo, ahi, 0,1,2,3,4,5,6,7,8,9,10,11,12,13,14,15);
      const __bf16* wp = wlds + ((size_t)((nt * 8 + kc) * 2) * 32 + lane) * 8;
      v8bf blo = *(const v8bf*)(wp);
      v8bf bhi = *(const v8bf*)(wp + 256);
      v16bf bm = __builtin_shufflevector(blo, bhi, 0,1,2,3,4,5,6,7,8,9,10,11,12,13,14,15);
      acc = __builtin_amdgcn_wmma_f32_16x16x32_bf16(
          false, a, false, bm, (short)0, acc, false, false);
    }

#pragma unroll
    for (int r = 0; r < 8; ++r) {
      int b = mt * 16 + r + 8 * half;
      float hv = tanhf(acc[r]);
      out_x[(b * RS + t) * RH + n] = hv;      // layer output (fp32)
      hn[b * HP + n] = (__bf16)hv;            // next-step h
      if (t == RS - 1) finals[b * RH + n] = hv;
    }
    __syncthreads();
    cur ^= 1;
  }
}

// ---------------------------------------------------------------------------
// Host-side orchestration. ws layout:
//   [0, 512KB)            : weight fragments (4 mats * 65536 bf16)
//   [512KB, 512KB+64MB)   : xp scratch (reused per layer)
//   [+64MB, +128MB)       : inter-layer activations x1
// d_out: [B*S*H] final-layer outputs, then [L*B*H] final hidden states.
// ---------------------------------------------------------------------------
extern "C" void kernel_launch(void* const* d_in, const int* in_sizes, int n_in,
                              void* d_out, int out_size, void* d_ws, size_t ws_size,
                              hipStream_t stream) {
  (void)in_sizes; (void)n_in; (void)out_size; (void)ws_size;
  const float* x_in = (const float*)d_in[0];
  const float* h0   = (const float*)d_in[1];
  const float* W_ih = (const float*)d_in[2];
  const float* W_hh = (const float*)d_in[3];
  const float* b_ih = (const float*)d_in[4];
  const float* b_hh = (const float*)d_in[5];

  float* out_x   = (float*)d_out;                               // [B,S,H]
  float* out_fin = (float*)d_out + (size_t)RB * RS * RH;        // [L,B,H]

  char* ws = (char*)d_ws;
  __bf16* wfrag = (__bf16*)ws;                                  // 4*65536 bf16
  float*  xp    = (float*)(ws + 524288);
  float*  x1    = (float*)(ws + 524288 + (size_t)RB * RS * RH * 4);

  const size_t lds_bytes = (size_t)65536 * 2 + (size_t)2 * RB * HP * 2;

  prep_weights_kernel<<<1024, 256, 0, stream>>>(W_ih, W_hh, wfrag);

  // layer 0
  proj_kernel<<<2048, 256, 0, stream>>>(x_in, wfrag + 0 * 65536, b_ih, b_hh, xp);
  scan_kernel<<<1, 1024, lds_bytes, stream>>>(xp, wfrag + 1 * 65536, h0, x1, out_fin);
  // layer 1
  proj_kernel<<<2048, 256, 0, stream>>>(x1, wfrag + 2 * 65536, b_ih + RH, b_hh + RH, xp);
  scan_kernel<<<1, 1024, lds_bytes, stream>>>(xp, wfrag + 3 * 65536, h0 + RB * RH,
                                              out_x, out_fin + RB * RH);
}
